// LinearAttention_29807073034926
// MI455X (gfx1250) — compile-verified, matrix operand order checked by reading, not run on hardware
//
#include <hip/hip_runtime.h>
#include <cstddef>
#include <cstdint>

// ---------------------------------------------------------------------------
// LinearAttention fused pipeline for gfx1250 (MI455X), wave32 + WMMA bf16
// + Tensor Data Mover staging (loads and stores).
//   b=32, C=128, H=W=64 -> n=4096, heads=4, dim_head=32, hidden=128, O=384
// ---------------------------------------------------------------------------

typedef __bf16 bf16;
typedef __attribute__((ext_vector_type(8)))  bf16  v8bf;
typedef __attribute__((ext_vector_type(16))) bf16  v16bf;
typedef __attribute__((ext_vector_type(8)))  float v8f;
typedef unsigned int u32x4 __attribute__((ext_vector_type(4)));
typedef int          i32x4 __attribute__((ext_vector_type(4)));
typedef int          i32x8 __attribute__((ext_vector_type(8)));

union Frag { v16bf v; v8bf h[2]; };

__device__ __forceinline__ v8f zero8() {
  v8f z = {0.f, 0.f, 0.f, 0.f, 0.f, 0.f, 0.f, 0.f};
  return z;
}

__device__ __forceinline__ v8f wmma_bf16(const Frag& a, const Frag& b, v8f c) {
  // D = A(16x32 bf16) x B(32x16 bf16) + C(16x16 f32)
  return __builtin_amdgcn_wmma_f32_16x16x32_bf16(
      /*neg_a=*/false, a.v, /*neg_b=*/false, b.v,
      /*c_mod=*/(short)0, c, /*reuse_a=*/false, /*reuse_b=*/false);
}

// A-style fragment load per ISA 7.12.2 (16-bit A 16x32): lanes 0-15 hold row
// M=lane with K runs {0..7, 16..23}; lanes 16-31 hold row M=lane-16 with runs
// {8..15, 24..31}. Same pattern serves B when memory is stored [col][K].
__device__ __forceinline__ Frag load_frag(const bf16* base, int row0, int ld, int k0) {
  const int lane = threadIdx.x & 31;
  const int hl = lane >> 4;
  const int m  = lane & 15;
  const bf16* p = base + (size_t)(row0 + m) * ld + (k0 + 8 * hl);
  Frag f;
  f.h[0] = *(const v8bf*)(p);
  f.h[1] = *(const v8bf*)(p + 16);
  return f;
}

#define BATCH 32
#define CCH   128
#define NPIX  4096
#define OQKV  384
#define HEADS 4
#define DHEAD 32
#define Q_SCALE 0.17677669529663687f  // 1/sqrt(32)

// ------------------------- Tensor Data Mover (TDM) -------------------------
#if defined(__has_builtin)
#if __has_builtin(__builtin_amdgcn_tensor_load_to_lds) && \
    __has_builtin(__builtin_amdgcn_s_wait_tensorcnt)
#define HAVE_TDM 1
#endif
#if __has_builtin(__builtin_amdgcn_tensor_store_from_lds) && \
    __has_builtin(__builtin_amdgcn_s_wait_tensorcnt)
#define HAVE_TDM_ST 1
#endif
#endif
#ifndef HAVE_TDM
#define HAVE_TDM 0
#endif
#ifndef HAVE_TDM_ST
#define HAVE_TDM_ST 0
#endif

#if HAVE_TDM || HAVE_TDM_ST
struct TdmDesc { u32x4 g0; i32x8 g1; };

// 2-D tile descriptor: tensor [tdim1][tdim0] (dim0 contiguous, stride0
// elements between rows), tile [tile1][tile0] at gaddr, LDS side contiguous
// at lds_addr. Bit layout per CDNA5 ISA.
__device__ __forceinline__ TdmDesc tdm_make_2d(unsigned lds_addr,
                                               const void* gaddr,
                                               unsigned dsz_code,  // 0=1B,1=2B,2=4B
                                               unsigned tdim0, unsigned tdim1,
                                               unsigned tile0, unsigned tile1,
                                               unsigned long long stride0) {
  const unsigned long long ga = (unsigned long long)(size_t)gaddr;
  TdmDesc d;
  d.g0.x = 1u;                                                // count=1 (valid)
  d.g0.y = lds_addr;                                          // lds_addr [63:32]
  d.g0.z = (unsigned)(ga & 0xffffffffu);                      // global_addr lo
  d.g0.w = (unsigned)((ga >> 32) & 0x1ffffffu) | (2u << 30);  // addr hi | type=2
  d.g1[0] = (int)(dsz_code << 16);                                        // data_size
  d.g1[1] = (int)((tdim0 & 0xffffu) << 16);                               // tdim0 lo
  d.g1[2] = (int)(((tdim0 >> 16) & 0xffffu) | ((tdim1 & 0xffffu) << 16)); // tdim0 hi|tdim1 lo
  d.g1[3] = (int)(((tdim1 >> 16) & 0xffffu) | ((tile0 & 0xffffu) << 16)); // tdim1 hi|tile0
  d.g1[4] = (int)(tile1 & 0xffffu);                                       // tile1 (tile2=0)
  d.g1[5] = (int)(unsigned)(stride0 & 0xffffffffu);                       // dim0 stride lo
  d.g1[6] = (int)(unsigned)((stride0 >> 32) & 0xffffu);                   // stride hi
  d.g1[7] = 0;
  return d;
}
#endif

#if HAVE_TDM
__device__ __forceinline__ void tdm_load_tile_2d(unsigned lds_addr, const void* gaddr,
                                                 unsigned dsz_code,
                                                 unsigned tdim0, unsigned tdim1,
                                                 unsigned tile0, unsigned tile1,
                                                 unsigned long long stride0) {
  TdmDesc d = tdm_make_2d(lds_addr, gaddr, dsz_code, tdim0, tdim1, tile0, tile1, stride0);
  i32x4 gz = {0, 0, 0, 0};
#if __clang_major__ >= 23
  i32x8 gz8 = {0, 0, 0, 0, 0, 0, 0, 0};
  __builtin_amdgcn_tensor_load_to_lds(d.g0, d.g1, gz, gz, gz8, 0);
#else
  __builtin_amdgcn_tensor_load_to_lds(d.g0, d.g1, gz, gz, 0);
#endif
}
#endif

#if HAVE_TDM_ST
__device__ __forceinline__ void tdm_store_tile_2d(unsigned lds_addr, const void* gaddr,
                                                  unsigned dsz_code,
                                                  unsigned tdim0, unsigned tdim1,
                                                  unsigned tile0, unsigned tile1,
                                                  unsigned long long stride0) {
  TdmDesc d = tdm_make_2d(lds_addr, gaddr, dsz_code, tdim0, tdim1, tile0, tile1, stride0);
  i32x4 gz = {0, 0, 0, 0};
#if __clang_major__ >= 23
  i32x8 gz8 = {0, 0, 0, 0, 0, 0, 0, 0};
  __builtin_amdgcn_tensor_store_from_lds(d.g0, d.g1, gz, gz, gz8, 0);
#else
  __builtin_amdgcn_tensor_store_from_lds(d.g0, d.g1, gz, gz, 0);
#endif
}
#endif

// --------------------------- K0: weight prep -------------------------------
__global__ __launch_bounds__(256) void k0_prep(const float* __restrict__ Wqkv,
                                               const float* __restrict__ Wout,
                                               bf16* __restrict__ wq,
                                               bf16* __restrict__ wo,
                                               float* __restrict__ bstats) {
  int t = blockIdx.x * 256 + threadIdx.x;
  if (t < OQKV * CCH) wq[t] = (bf16)Wqkv[t];
  if (t < CCH * CCH)  wo[t] = (bf16)Wout[t];
  if (t < 2 * BATCH)  bstats[t] = 0.f;
}

// ------------------- K1: qkv = W_qkv @ x  (bf16 WMMA) ----------------------
// grid (32 ntiles, 32 batch), 256 thr. Each WG: all 384 rows x 128 columns.
// x tile in via TDM; qkv tile staged in LDS, out via TDM store.
__global__ __launch_bounds__(256) void k1_qkv(const float* __restrict__ x,
                                              const bf16* __restrict__ Wq,
                                              bf16* __restrict__ qkv) {
  const int nt = blockIdx.x;
  const int b  = blockIdx.y;
  __shared__ union {
    float xstage[128][128];   // TDM landing zone: x tile [c][n], f32, 64KB
    bf16  dstage[OQKV][128];  // qkv result staging [o][n], bf16, 96KB
  } su;
  __shared__ bf16 xT[128][136];  // [n][k] transposed bf16, padded

  const float* xb = x + (size_t)b * CCH * NPIX + (size_t)nt * 128;

#if HAVE_TDM
  if (threadIdx.x == 0) {
    tdm_load_tile_2d((unsigned)(size_t)&su.xstage[0][0], (const void*)xb,
                     /*dsz=4B*/ 2u, NPIX, CCH, 128u, 128u, NPIX);
    __builtin_amdgcn_s_wait_tensorcnt(0);
  }
  __syncthreads();
  for (int idx = threadIdx.x * 4; idx < 128 * 128; idx += 256 * 4) {
    const int c = idx >> 7, j = idx & 127;
    const float4 f = *(const float4*)&su.xstage[c][j];
    xT[j + 0][c] = (bf16)f.x;
    xT[j + 1][c] = (bf16)f.y;
    xT[j + 2][c] = (bf16)f.z;
    xT[j + 3][c] = (bf16)f.w;
  }
#else
  for (int idx = threadIdx.x * 4; idx < 128 * 128; idx += 256 * 4) {
    const int c = idx >> 7, j = idx & 127;
    const float4 f = *(const float4*)(xb + (size_t)c * NPIX + j);
    xT[j + 0][c] = (bf16)f.x;
    xT[j + 1][c] = (bf16)f.y;
    xT[j + 2][c] = (bf16)f.z;
    xT[j + 3][c] = (bf16)f.w;
  }
#endif
  __syncthreads();

  const int wave = threadIdx.x >> 5;
  const int lane = threadIdx.x & 31;
  const int hl = lane >> 4, m = lane & 15;

  // Preload A fragments: 3 row strips (rows 16*(wave+8s)) x 4 K-steps
  Frag a[3][4];
#pragma unroll
  for (int s = 0; s < 3; ++s)
#pragma unroll
    for (int kk = 0; kk < 4; ++kk)
      a[s][kk] = load_frag(Wq, (wave + 8 * s) * 16, CCH, kk * 32);

  // two column tiles at a time -> 6 independent WMMAs per K-step (hides the
  // 5-slot WMMA->WMMA accumulator hazard without v_nop padding)
#pragma unroll
  for (int cp = 0; cp < 4; ++cp) {
    v8f acc[2][3] = {{zero8(), zero8(), zero8()}, {zero8(), zero8(), zero8()}};
#pragma unroll
    for (int kk = 0; kk < 4; ++kk) {
      Frag bf0 = load_frag(&xT[0][0], (2 * cp) * 16, 136, kk * 32);
      Frag bf1 = load_frag(&xT[0][0], (2 * cp + 1) * 16, 136, kk * 32);
#pragma unroll
      for (int s = 0; s < 3; ++s) acc[0][s] = wmma_bf16(a[s][kk], bf0, acc[0][s]);
#pragma unroll
      for (int s = 0; s < 3; ++s) acc[1][s] = wmma_bf16(a[s][kk], bf1, acc[1][s]);
    }
#pragma unroll
    for (int hc = 0; hc < 2; ++hc) {
      const int ct = 2 * cp + hc;
#pragma unroll
      for (int s = 0; s < 3; ++s) {
        const int row0 = (wave + 8 * s) * 16 + 8 * hl;
#pragma unroll
        for (int r = 0; r < 8; ++r)
          su.dstage[row0 + r][ct * 16 + m] = (bf16)acc[hc][s][r];
      }
    }
  }
  __syncthreads();

  bf16* outbase = qkv + (size_t)b * OQKV * NPIX + (size_t)nt * 128;
#if HAVE_TDM_ST
  if (threadIdx.x == 0) {
    // one DMA: 384x128 bf16 tile, LDS contiguous -> global rows stride 4096
    tdm_store_tile_2d((unsigned)(size_t)&su.dstage[0][0], (const void*)outbase,
                      /*dsz=2B*/ 1u, NPIX, OQKV, 128u, OQKV, NPIX);
    __builtin_amdgcn_s_wait_tensorcnt(0);
  }
#else
  for (int idx = threadIdx.x * 8; idx < OQKV * 128; idx += 256 * 8) {
    const int row = idx >> 7, j = idx & 127;
    *(v8bf*)(outbase + (size_t)row * NPIX + j) = *(const v8bf*)&su.dstage[row][j];
  }
#endif
}

// -------------- K2: q softmax over d (per pixel), fold scale ---------------
__global__ __launch_bounds__(256) void k2_qsoftmax(bf16* __restrict__ qkv) {
  const int n = blockIdx.x * 256 + threadIdx.x;
  const int h = blockIdx.y, b = blockIdx.z;
  bf16* q = qkv + ((size_t)b * OQKV + (size_t)h * DHEAD) * NPIX + n;
  float v[DHEAD];
  float mx = -3.0e38f;
#pragma unroll
  for (int d = 0; d < DHEAD; ++d) {
    v[d] = (float)q[(size_t)d * NPIX];
    mx = fmaxf(mx, v[d]);
  }
  float s = 0.f;
#pragma unroll
  for (int d = 0; d < DHEAD; ++d) {
    v[d] = __expf(v[d] - mx);
    s += v[d];
  }
  const float inv = Q_SCALE / s;
#pragma unroll
  for (int d = 0; d < DHEAD; ++d) q[(size_t)d * NPIX] = (bf16)(v[d] * inv);
}

// ------------- K3: per-row k stats (max, 1/sum exp) over n=4096 ------------
__global__ __launch_bounds__(256) void k3_kstats(const bf16* __restrict__ qkv,
                                                 float* __restrict__ kstats) {
  const int rid = blockIdx.x;          // b*128 + h*32 + d
  const int b = rid >> 7, hd = rid & 127;
  const bf16* krow = qkv + ((size_t)b * OQKV + CCH + hd) * NPIX;
  __shared__ float red[256];
  const int tid = threadIdx.x;

  float mx = -3.0e38f;
  for (int i = tid; i < NPIX; i += 256) {
    if (i + 2048 < NPIX) __builtin_prefetch(krow + i + 2048, 0, 0);
    mx = fmaxf(mx, (float)krow[i]);
  }
  red[tid] = mx;
  __syncthreads();
  for (int ofs = 128; ofs > 0; ofs >>= 1) {
    if (tid < ofs) red[tid] = fmaxf(red[tid], red[tid + ofs]);
    __syncthreads();
  }
  mx = red[0];
  __syncthreads();

  float s = 0.f;
  for (int i = tid; i < NPIX; i += 256) s += __expf((float)krow[i] - mx);
  red[tid] = s;
  __syncthreads();
  for (int ofs = 128; ofs > 0; ofs >>= 1) {
    if (tid < ofs) red[tid] += red[tid + ofs];
    __syncthreads();
  }
  if (tid == 0) {
    kstats[2 * rid]     = mx;
    kstats[2 * rid + 1] = 1.0f / red[0];
  }
}

// ---- K4: context[d,e] = sum_n softmax(k)[d,n] v[e,n]; store ctx^T bf16 ----
__global__ __launch_bounds__(256) void k4_context(const bf16* __restrict__ qkv,
                                                  const float* __restrict__ kstats,
                                                  bf16* __restrict__ ctxT) {
  const int b = blockIdx.x >> 2, h = blockIdx.x & 3;
  const bf16* kbase = qkv + ((size_t)b * OQKV + CCH + h * DHEAD) * NPIX;
  const bf16* vbase = qkv + ((size_t)b * OQKV + 2 * CCH + h * DHEAD) * NPIX;

  const int wave = threadIdx.x >> 5;
  const int lane = threadIdx.x & 31;
  const int hl = lane >> 4, m = lane & 15;

  float kmx[2];
#pragma unroll
  for (int s = 0; s < 2; ++s)
    kmx[s] = kstats[2 * (((b * 4 + h) * DHEAD) + s * 16 + m)];

  v8f acc[2][2] = {{zero8(), zero8()}, {zero8(), zero8()}};
  for (int it = 0; it < 16; ++it) {
    const int n0 = wave * 512 + it * 32;
    Frag afr[2], bfr[2];
#pragma unroll
    for (int s = 0; s < 2; ++s) {
      const bf16* p = kbase + (size_t)(s * 16 + m) * NPIX + n0 + 8 * hl;
      v8bf r0 = *(const v8bf*)(p);
      v8bf r1 = *(const v8bf*)(p + 16);
#pragma unroll
      for (int j = 0; j < 8; ++j) {
        afr[s].h[0][j] = (bf16)__expf((float)r0[j] - kmx[s]);
        afr[s].h[1][j] = (bf16)__expf((float)r1[j] - kmx[s]);
      }
      bfr[s] = load_frag(vbase, s * 16, NPIX, n0);  // v: [e][n] == B [col][K]
    }
#pragma unroll
    for (int dt = 0; dt < 2; ++dt)
#pragma unroll
      for (int et = 0; et < 2; ++et)
        acc[dt][et] = wmma_bf16(afr[dt], bfr[et], acc[dt][et]);
  }

  __shared__ float red[8][4][256];
#pragma unroll
  for (int dt = 0; dt < 2; ++dt)
#pragma unroll
    for (int et = 0; et < 2; ++et)
#pragma unroll
      for (int r = 0; r < 8; ++r)
        red[wave][dt * 2 + et][lane * 8 + r] = acc[dt][et][r];
  __syncthreads();

  if (wave < 4) {
    const int dt = wave >> 1, et = wave & 1;
#pragma unroll
    for (int r = 0; r < 8; ++r) {
      float s = 0.f;
#pragma unroll
      for (int w = 0; w < 8; ++w) s += red[w][wave][lane * 8 + r];
      const int d = dt * 16 + 8 * hl + r;      // row of context
      const int e = et * 16 + m;               // col of context
      const float inv = kstats[2 * (((b * 4 + h) * DHEAD) + d) + 1];
      ctxT[(((size_t)(b * 4 + h)) * DHEAD + e) * DHEAD + d] = (bf16)(s * inv);
    }
  }
}

// ---- K5: out = ctx^T @ q per head, then y = W_out @ out + b, GN partials --
__global__ __launch_bounds__(256) void k5_out(const bf16* __restrict__ qkv,
                                              const bf16* __restrict__ ctxT,
                                              const bf16* __restrict__ Wo,
                                              const float* __restrict__ bias,
                                              float* __restrict__ y,
                                              float* __restrict__ bstats) {
  const int nt = blockIdx.x;
  const int b  = blockIdx.y;
  __shared__ bf16 qT[128][136];  // [n][d_all]
  __shared__ union {
    bf16 qstage[128][128];  // TDM landing zone (dead after transpose)
    bf16 oT[128][136];      // [n][c] intermediate
  } sv;

  const bf16* qbase = qkv + (size_t)b * OQKV * NPIX + (size_t)nt * 128;
#if HAVE_TDM
  if (threadIdx.x == 0) {
    tdm_load_tile_2d((unsigned)(size_t)&sv.qstage[0][0], (const void*)qbase,
                     /*dsz=2B*/ 1u, NPIX, CCH, 128u, 128u, NPIX);
    __builtin_amdgcn_s_wait_tensorcnt(0);
  }
  __syncthreads();
  for (int idx = threadIdx.x * 8; idx < 128 * 128; idx += 256 * 8) {
    const int d = idx >> 7, j = idx & 127;
    v8bf r = *(const v8bf*)&sv.qstage[d][j];
#pragma unroll
    for (int t = 0; t < 8; ++t) qT[j + t][d] = r[t];
  }
#else
  for (int idx = threadIdx.x * 8; idx < 128 * 128; idx += 256 * 8) {
    const int d = idx >> 7, j = idx & 127;
    v8bf r = *(const v8bf*)(qbase + (size_t)d * NPIX + j);
#pragma unroll
    for (int t = 0; t < 8; ++t) qT[j + t][d] = r[t];
  }
#endif
  __syncthreads();

  const int wave = threadIdx.x >> 5;
  const int lane = threadIdx.x & 31;
  const int hl = lane >> 4, m = lane & 15;

  // ----- stage 1: out rows c = 16*wave .. 16*wave+15 (head h = wave/2) -----
  {
    const int h = wave >> 1;
    const int ebase = (wave & 1) * 16;
    Frag actx = load_frag(ctxT + (size_t)(b * 4 + h) * DHEAD * DHEAD,
                          ebase, DHEAD, 0);  // rows e, K = d(32)
#pragma unroll
    for (int t2 = 0; t2 < 8; ++t2) {
      Frag bq = load_frag(&qT[0][0], t2 * 16, 136, h * DHEAD);
      v8f acc = wmma_bf16(actx, bq, zero8());
      __align__(16) bf16 tmp[8];
#pragma unroll
      for (int r = 0; r < 8; ++r) tmp[r] = (bf16)acc[r];
      *(v8bf*)(&sv.oT[t2 * 16 + m][16 * wave + 8 * hl]) = *(const v8bf*)tmp;
    }
  }
  __syncthreads();

  // ----- stage 2: y rows o = 16*wave .. +15, K = 128 over c ----------------
  Frag aw[4];
#pragma unroll
  for (int kk = 0; kk < 4; ++kk)
    aw[kk] = load_frag(Wo, 16 * wave, CCH, kk * 32);
  float b0[8];
#pragma unroll
  for (int r = 0; r < 8; ++r) b0[r] = bias[16 * wave + 8 * hl + r];

  float s1 = 0.f, s2 = 0.f;
#pragma unroll
  for (int cg = 0; cg < 2; ++cg) {  // 4 column tiles in flight
    v8f acc[4] = {zero8(), zero8(), zero8(), zero8()};
#pragma unroll
    for (int kk = 0; kk < 4; ++kk) {
#pragma unroll
      for (int t = 0; t < 4; ++t) {
        Frag bo = load_frag(&sv.oT[0][0], (cg * 4 + t) * 16, 136, kk * 32);
        acc[t] = wmma_bf16(aw[kk], bo, acc[t]);
      }
    }
#pragma unroll
    for (int t = 0; t < 4; ++t) {
      const int ct = cg * 4 + t;
      const size_t col = (size_t)nt * 128 + ct * 16 + m;
      float* yp = y + ((size_t)b * CCH + 16 * wave + 8 * hl) * NPIX + col;
#pragma unroll
      for (int r = 0; r < 8; ++r) {
        const float vv = acc[t][r] + b0[r];
        yp[(size_t)r * NPIX] = vv;
        s1 += vv;
        s2 += vv * vv;
      }
    }
  }

  __shared__ float rs1[256], rs2[256];
  const int tid = threadIdx.x;
  rs1[tid] = s1;
  rs2[tid] = s2;
  __syncthreads();
  for (int ofs = 128; ofs > 0; ofs >>= 1) {
    if (tid < ofs) {
      rs1[tid] += rs1[tid + ofs];
      rs2[tid] += rs2[tid + ofs];
    }
    __syncthreads();
  }
  if (tid == 0) {
    atomicAdd(&bstats[2 * b], rs1[0]);
    atomicAdd(&bstats[2 * b + 1], rs2[0]);
  }
}

// --------------------- K6: finalize per-batch mean/rstd --------------------
__global__ void k6_stats(const float* __restrict__ bstats,
                         float* __restrict__ musd) {
  const int b = threadIdx.x;
  if (b < BATCH) {
    const float N = (float)(CCH * NPIX);
    const float mu = bstats[2 * b] / N;
    const float var = bstats[2 * b + 1] / N - mu * mu;
    musd[2 * b] = mu;
    musd[2 * b + 1] = rsqrtf(var + 1e-5f);
  }
}

// -------------------- K7: GroupNorm apply, in place on y -------------------
__global__ __launch_bounds__(256) void k7_norm(float* __restrict__ y,
                                               const float* __restrict__ musd,
                                               const float* __restrict__ gamma,
                                               const float* __restrict__ beta) {
  const size_t i = ((size_t)blockIdx.x * 256 + threadIdx.x) * 4;
  const int b = (int)(i >> 19);          // 128*4096 = 2^19 per batch
  const int c = (int)((i >> 12) & 127);  // 4096 = 2^12 per channel
  const float mu = musd[2 * b], rs = musd[2 * b + 1];
  const float g = gamma[c] * rs;
  const float bt = beta[c] - mu * g;
  float4 v = *(float4*)(y + i);
  v.x = v.x * g + bt;
  v.y = v.y * g + bt;
  v.z = v.z * g + bt;
  v.w = v.w * g + bt;
  *(float4*)(y + i) = v;
}

// ---------------------------------------------------------------------------
extern "C" void kernel_launch(void* const* d_in, const int* in_sizes, int n_in,
                              void* d_out, int out_size, void* d_ws, size_t ws_size,
                              hipStream_t stream) {
  const float* x     = (const float*)d_in[0];
  const float* Wqkv  = (const float*)d_in[1];
  const float* Wout  = (const float*)d_in[2];
  const float* bout  = (const float*)d_in[3];
  const float* gamma = (const float*)d_in[4];
  const float* beta  = (const float*)d_in[5];
  float* y = (float*)d_out;  // staging for pre-norm y, then normalized in place

  char* ws = (char*)d_ws;
  bf16*  wq     = (bf16*)(ws + 0);        // 384*128 bf16   = 98304 B
  bf16*  wo     = (bf16*)(ws + 98304);    // 128*128 bf16   = 32768 B
  bf16*  ctxT   = (bf16*)(ws + 131072);   // 128*32*32 bf16 = 262144 B
  float* kstats = (float*)(ws + 393216);  // 4096*2 f32     = 32768 B
  float* bstats = (float*)(ws + 425984);  // 32*2 f32
  float* musd   = (float*)(ws + 426240);  // 32*2 f32
  bf16*  qkv    = (bf16*)(ws + 524288);   // 32*384*4096 bf16 = 96 MiB

  k0_prep    <<<192, 256, 0, stream>>>(Wqkv, Wout, wq, wo, bstats);
  k1_qkv     <<<dim3(32, 32), 256, 0, stream>>>(x, wq, qkv);
  k2_qsoftmax<<<dim3(16, 4, 32), 256, 0, stream>>>(qkv);
  k3_kstats  <<<4096, 256, 0, stream>>>(qkv, kstats);
  k4_context <<<128, 256, 0, stream>>>(qkv, kstats, ctxT);
  k5_out     <<<dim3(32, 32), 256, 0, stream>>>(qkv, ctxT, wo, bout, y, bstats);
  k6_stats   <<<1, 32, 0, stream>>>(bstats, musd);
  k7_norm    <<<16384, 256, 0, stream>>>(y, musd, gamma, beta);
}
